// BandSplit_20607253086824
// MI455X (gfx1250) — compile-verified, matrix operand order checked by reading, not run on hardware
//
#include <hip/hip_runtime.h>

// ---------------------------------------------------------------------------
// BandSplit fused kernel for gfx1250 (MI455X, wave32, WMMA).
//
// gather -> [per-band GEMM1 (BTxd_in)x(d_inx128)] -> +bias -> (LDS)
//        -> [per-band GEMM2 (BTx128)x(128xd_in)]  -> +bias -> mask
//        -> scatter-add -> /ola_window
//
// fp32 inputs are split into bf16 hi/lo pairs; each fp32 MAC is emulated with
// 3x v_wmma_f32_16x16x32_bf16 (hi*hi + hi*lo + lo*hi), ~2^-17 rel error.
// ---------------------------------------------------------------------------

typedef __bf16 v16bf __attribute__((ext_vector_type(16)));
typedef float  v8f   __attribute__((ext_vector_type(8)));

#define BB 4      // batch
#define CC 4      // channels in
#define TT 512    // time
#define FF 1025   // freq bins
#define KB 256    // bands
#define DH 128    // hidden (C_OUT)
#define BT (BB*TT)        // 2048 rows
#define MCHUNK 64         // rows per workgroup

// Map K index within a 32-deep slice -> (lane-half, element j) per the CDNA5
// 16-bit A-matrix VGPR layout (ISA 7.12.2).  B uses the mirrored layout.
__device__ __forceinline__ void kmap(int kk, int &half, int &j) {
    if (kk < 8)       { half = 0; j = kk;      }
    else if (kk < 16) { half = 1; j = kk - 8;  }
    else if (kk < 24) { half = 0; j = kk - 8;  }
    else              { half = 1; j = kk - 16; }
}

__device__ __forceinline__ void bsplit(float v, __bf16 &hi, __bf16 &lo) {
    hi = (__bf16)v;
    lo = (__bf16)(v - (float)hi);
}

__device__ __forceinline__ v8f wmma_bf16(v16bf a, v16bf b, v8f c) {
    return __builtin_amdgcn_wmma_f32_16x16x32_bf16(false, a, false, b,
                                                   (short)0, c, false, false);
}

extern "C" __global__ __launch_bounds__(256)
void bandsplit_fused(const float* __restrict__ x,      // (B,C,T,F)
                     const float* __restrict__ w1,     // (K,d_in,128)
                     const float* __restrict__ b1,     // (K,128)
                     const float* __restrict__ w2,     // (K,128,d_in)
                     const float* __restrict__ b2,     // (K,d_in)
                     const float* __restrict__ mask,   // (K*W)
                     const int*   __restrict__ fidx,   // (K*W)
                     float*       __restrict__ out,    // (B,C,T,F), pre-zeroed
                     int W, int ks1, int nt2)
{
    extern __shared__ __bf16 smem[];
    const int d_in = 4 * W;
    const int k    = blockIdx.y;           // band
    const int m0   = blockIdx.x * MCHUNK;  // first row of chunk
    const int tid  = threadIdx.x;
    const int lane = tid & 31;
    const int wv   = tid >> 5;             // wave id 0..7

    // LDS regions, each "tile-slice" = 32 lanes * 16 bf16 = 512 elements.
    const int szA  = 4   * ks1 * 512;      // A fragments (4 M-tiles)
    const int szW1 = 8   * ks1 * 512;      // W1 fragments (8 N-tiles)
    const int szH  = 4   * 4   * 512;      // H fragments (4 M-tiles x 4 K-slices)
    const int szW2 = nt2 * 4   * 512;      // W2 fragments
    __bf16 *lAhi  = smem;
    __bf16 *lAlo  = lAhi  + szA;
    __bf16 *lW1hi = lAlo  + szA;
    __bf16 *lW1lo = lW1hi + szW1;
    __bf16 *lHhi  = lW1lo + szW1;
    __bf16 *lHlo  = lHhi  + szH;
    __bf16 *lW2hi = lHlo  + szH;
    __bf16 *lW2lo = lW2hi + szW2;

    const int kp1 = ks1 * 32;              // padded d_in

    // ---- cooperative fill: gathered A chunk (64 x kp1), swizzled ----
    for (int e = tid; e < MCHUNK * kp1; e += 256) {
        int m = e / kp1, i = e - m * kp1;
        float v = 0.f;
        if (i < d_in) {
            int wi = i >> 2, c = i & 3;
            int idx = k * W + wi;
            int f   = fidx[idx];
            int rg  = m0 + m;
            int b   = rg >> 9, t = rg & (TT - 1);
            v = x[(((b * CC + c) * TT) + t) * FF + f] * mask[idx];
        }
        int half, j; kmap(i & 31, half, j);
        int off = (((m >> 4) * ks1 + (i >> 5)) * 32 + ((m & 15) + 16 * half)) * 16 + j;
        __bf16 hi, lo; bsplit(v, hi, lo);
        lAhi[off] = hi; lAlo[off] = lo;
    }
    // ---- W1 (kp1 x 128), swizzled B layout ----
    for (int e = tid; e < kp1 * DH; e += 256) {
        int i = e / DH, o = e - i * DH;
        float v = (i < d_in) ? w1[(k * d_in + i) * DH + o] : 0.f;
        int half, j; kmap(i & 31, half, j);
        int off = (((o >> 4) * ks1 + (i >> 5)) * 32 + ((o & 15) + 16 * half)) * 16 + j;
        __bf16 hi, lo; bsplit(v, hi, lo);
        lW1hi[off] = hi; lW1lo[off] = lo;
    }
    // ---- W2 (128 x nt2*16), swizzled B layout ----
    const int np2 = nt2 * 16;
    for (int e = tid; e < DH * np2; e += 256) {
        int d = e / np2, jj = e - d * np2;
        float v = (jj < d_in) ? w2[(k * DH + d) * d_in + jj] : 0.f;
        int half, j; kmap(d & 31, half, j);
        int off = (((jj >> 4) * 4 + (d >> 5)) * 32 + ((jj & 15) + 16 * half)) * 16 + j;
        __bf16 hi, lo; bsplit(v, hi, lo);
        lW2hi[off] = hi; lW2lo[off] = lo;
    }
    __syncthreads();

    // ---- stage 1: H = A * W1  (each wave: 1 M-tile x 4 N-tiles) ----
    const int mt = wv & 3;        // M-tile 0..3
    const int ng = wv >> 1 >> 1;  // N-group 0..1
    v8f acc[4];
    #pragma unroll
    for (int q = 0; q < 4; ++q)
        #pragma unroll
        for (int r = 0; r < 8; ++r) acc[q][r] = 0.f;

    for (int ks = 0; ks < ks1; ++ks) {
        v16bf ahi = *(const v16bf*)&lAhi[((mt * ks1 + ks) * 32 + lane) * 16];
        v16bf alo = *(const v16bf*)&lAlo[((mt * ks1 + ks) * 32 + lane) * 16];
        #pragma unroll
        for (int q = 0; q < 4; ++q) {
            int nt = ng * 4 + q;
            v16bf bhi = *(const v16bf*)&lW1hi[((nt * ks1 + ks) * 32 + lane) * 16];
            v16bf blo = *(const v16bf*)&lW1lo[((nt * ks1 + ks) * 32 + lane) * 16];
            acc[q] = wmma_bf16(ahi, bhi, acc[q]);
            acc[q] = wmma_bf16(ahi, blo, acc[q]);
            acc[q] = wmma_bf16(alo, bhi, acc[q]);
        }
    }

    // ---- +pre_bias, write H straight into swizzled A-fragment layout ----
    const int halfC = lane >> 4;
    #pragma unroll
    for (int q = 0; q < 4; ++q) {
        int n = (ng * 4 + q) * 16 + (lane & 15);   // hidden index 0..127
        float bias = b1[k * DH + n];
        int ks2 = n >> 5;
        int half, j; kmap(n & 31, half, j);
        #pragma unroll
        for (int r = 0; r < 8; ++r) {
            int m = mt * 16 + r + 8 * halfC;
            float v = acc[q][r] + bias;
            int off = (((m >> 4) * 4 + ks2) * 32 + ((m & 15) + 16 * half)) * 16 + j;
            __bf16 hi, lo; bsplit(v, hi, lo);
            lHhi[off] = hi; lHlo[off] = lo;
        }
    }
    __syncthreads();

    // ---- stage 2: Y = H * W2, masked scatter-add to output ----
    for (int t2 = ng; t2 < nt2; t2 += 2) {
        v8f c2;
        #pragma unroll
        for (int r = 0; r < 8; ++r) c2[r] = 0.f;
        #pragma unroll
        for (int ks = 0; ks < 4; ++ks) {
            v16bf ahi = *(const v16bf*)&lHhi[((mt * 4 + ks) * 32 + lane) * 16];
            v16bf alo = *(const v16bf*)&lHlo[((mt * 4 + ks) * 32 + lane) * 16];
            v16bf bhi = *(const v16bf*)&lW2hi[((t2 * 4 + ks) * 32 + lane) * 16];
            v16bf blo = *(const v16bf*)&lW2lo[((t2 * 4 + ks) * 32 + lane) * 16];
            c2 = wmma_bf16(ahi, bhi, c2);
            c2 = wmma_bf16(ahi, blo, c2);
            c2 = wmma_bf16(alo, bhi, c2);
        }
        int jcol = t2 * 16 + (lane & 15);
        if (jcol < d_in) {
            int wi = jcol >> 2, c = jcol & 3;
            int idx = k * W + wi;
            float mk = mask[idx];
            if (mk != 0.f) {
                int f = fidx[idx];
                float bias = b2[k * d_in + jcol];
                #pragma unroll
                for (int r = 0; r < 8; ++r) {
                    int m  = mt * 16 + r + 8 * halfC;
                    int rg = m0 + m;
                    int b  = rg >> 9, t = rg & (TT - 1);
                    float val = (c2[r] + bias) * mk;
                    __hip_atomic_fetch_add(&out[(((b * CC + c) * TT) + t) * FF + f],
                                           val, __ATOMIC_RELAXED,
                                           __HIP_MEMORY_SCOPE_AGENT);
                }
            }
        }
    }
}

extern "C" __global__ void zero_out_k(float* __restrict__ out, int n) {
    int i = blockIdx.x * 256 + threadIdx.x;
    if (i < n) out[i] = 0.f;
}

extern "C" __global__ void ola_scale_k(float* __restrict__ out,
                                       const float* __restrict__ ola, int n) {
    int i = blockIdx.x * 256 + threadIdx.x;
    if (i < n) out[i] = out[i] / ola[i % FF];
}

extern "C" void kernel_launch(void* const* d_in, const int* in_sizes, int n_in,
                              void* d_out, int out_size, void* d_ws, size_t ws_size,
                              hipStream_t stream) {
    const float* x    = (const float*)d_in[0];
    const float* w1   = (const float*)d_in[1];
    const float* b1   = (const float*)d_in[2];
    const float* w2   = (const float*)d_in[3];
    const float* b2   = (const float*)d_in[4];
    const float* mask = (const float*)d_in[5];
    const float* ola  = (const float*)d_in[6];
    const int*   fidx = (const int*)d_in[7];
    float* out = (float*)d_out;

    const int W    = in_sizes[7] / KB;       // band width (runtime)
    const int dinf = 4 * W;
    const int ks1  = (dinf + 31) / 32;       // K-slices, stage 1
    const int nt2  = (dinf + 15) / 16;       // N-tiles, stage 2
    const size_t lds = (size_t)(4 * ks1 + 8 * ks1 + 16 + 4 * nt2)
                       * 512 * sizeof(__bf16) * 2 /* hi+lo */;

    hipFuncSetAttribute(reinterpret_cast<const void*>(bandsplit_fused),
                        hipFuncAttributeMaxDynamicSharedMemorySize, (int)lds);

    zero_out_k<<<(out_size + 255) / 256, 256, 0, stream>>>(out, out_size);

    dim3 grid(BT / MCHUNK, KB);              // 32 row-chunks x 256 bands
    bandsplit_fused<<<grid, 256, lds, stream>>>(x, w1, b1, w2, b2, mask, fidx,
                                                out, W, ks1, nt2);

    ola_scale_k<<<(out_size + 255) / 256, 256, 0, stream>>>(out, ola, out_size);
}
